// PREDR_45741401702576
// MI455X (gfx1250) — matrix operand order repaired and checked by reading, not compiled, not run on hardware
//
#include <hip/hip_runtime.h>
#include <hip/hip_bf16.h>
#include <math.h>

typedef _Float16 v8h  __attribute__((ext_vector_type(8)));
typedef _Float16 v16h __attribute__((ext_vector_type(16)));
typedef float    v8f  __attribute__((ext_vector_type(8)));

#define HDIM0 128
#define HDIM1 64
#define NROW  1024

// Load 16 halves (two contiguous 16B runs) into one WMMA fragment register set.
__device__ __forceinline__ v16h ld16(const _Float16* p, int off0, int off1) {
    v8h lo = *(const v8h*)(p + off0);
    v8h hi = *(const v8h*)(p + off1);
    return __builtin_shufflevector(lo, hi, 0,1,2,3,4,5,6,7,8,9,10,11,12,13,14,15);
}

// ---------------------------------------------------------------------------
// Stage 1a: hl = X_l @ W0[:5], hr' = X_r @ W0[5:] + b0  (f32 -> f16), both branches.
// 4 arrays of 1024x128. Trivial cost.
// ---------------------------------------------------------------------------
__global__ void stage1_h_kernel(const float* __restrict__ Xch, const float* __restrict__ Xg,
                                const float* __restrict__ Xd,
                                const float* __restrict__ Wchg0, const float* __restrict__ bchg0,
                                const float* __restrict__ Wdg0,  const float* __restrict__ bdg0,
                                _Float16* __restrict__ HLch, _Float16* __restrict__ HRbg,
                                _Float16* __restrict__ HLg,  _Float16* __restrict__ HRbd)
{
    int idx = blockIdx.x * blockDim.x + threadIdx.x;     // 4 * 1024 * 128 threads
    int which = idx >> 17;
    int rem   = idx & 131071;
    int row   = rem >> 7;
    int k     = rem & 127;
    const float* X; const float* W; float bias = 0.f; _Float16* O; int wo;
    switch (which) {
        case 0:  X = Xch; W = Wchg0; wo = 0; O = HLch; break;
        case 1:  X = Xg;  W = Wchg0; wo = 5; O = HRbg; bias = bchg0[k]; break;
        case 2:  X = Xg;  W = Wdg0;  wo = 0; O = HLg;  break;
        default: X = Xd;  W = Wdg0;  wo = 5; O = HRbd; bias = bdg0[k]; break;
    }
    float s = bias;
#pragma unroll
    for (int j = 0; j < 5; ++j)
        s = fmaf(X[row * 5 + j], W[(wo + j) * HDIM0 + k], s);
    O[row * HDIM0 + k] = (_Float16)s;
}

// ---------------------------------------------------------------------------
// Stage 1b: W1 (128x64 f32) -> W1T (64x128 f16) for both branches, so the
// B-fragment (column of W1) becomes contiguous per lane.
// ---------------------------------------------------------------------------
__global__ void stage1_w_kernel(const float* __restrict__ Wchg1, const float* __restrict__ Wdg1,
                                _Float16* __restrict__ W1Tchg, _Float16* __restrict__ W1Tdg)
{
    int idx = blockIdx.x * blockDim.x + threadIdx.x;     // 2 * 64 * 128 threads
    int which = idx >> 13;
    int rem   = idx & 8191;
    int n = rem >> 7;
    int k = rem & 127;
    const float* W = which ? Wdg1 : Wchg1;
    _Float16*    O = which ? W1Tdg : W1Tchg;
    O[n * HDIM0 + k] = (_Float16)W[k * HDIM1 + n];
}

// ---------------------------------------------------------------------------
// Stage 2: per-pair MLP via WMMA.
// Block = 8 waves; block tile = 16 l's x 64 r's; each wave: M = 16 l's, 8 r's.
// Per r: build h = relu(hl + hr') directly in 16x32 f16 A-fragment layout,
// 16x v_wmma_f32_16x16x32_f16 against resident W1 B-fragments, then the
// Dense(1) dot + cross-lane reduce + elu in registers.
// ---------------------------------------------------------------------------
__global__ __launch_bounds__(256) void pair_mlp_kernel(
    const _Float16* __restrict__ HL, const _Float16* __restrict__ HRb,
    const _Float16* __restrict__ W1T, const float* __restrict__ b1,
    const float* __restrict__ Wr, const float* __restrict__ br,
    float* __restrict__ Aout, _Float16* __restrict__ Af16, int transposeF16)
{
    const int tid  = threadIdx.x;
    const int lane = tid & 31;
    const int wave = tid >> 5;
    const int m    = lane & 15;        // A row / C col within tile
    const int hi   = lane >> 4;        // half-wave select
    const int l0   = blockIdx.x * 16;
    const int rbase = blockIdx.y * 64 + wave * 8;

    // Preload hl A-fragments for the 16 l rows (reused across all 8 r's).
    v16h hlv[4];
    const _Float16* hlrow = HL + (l0 + m) * HDIM0;
#pragma unroll
    for (int kt = 0; kt < 4; ++kt) {
        int k0 = 32 * kt + 8 * hi;
        hlv[kt] = ld16(hlrow, k0, k0 + 16);
    }
    // Preload all 16 W1 B-fragments (lane holds column n = 16*nt + m, two K runs of 8).
    v16h bfrag[4][4];
#pragma unroll
    for (int kt = 0; kt < 4; ++kt)
#pragma unroll
        for (int nt = 0; nt < 4; ++nt) {
            const _Float16* bp = W1T + (nt * 16 + m) * HDIM0 + 32 * kt + 8 * hi;
            bfrag[kt][nt] = ld16(bp, 0, 16);
        }
    float b1v[4], wrv[4];
#pragma unroll
    for (int nt = 0; nt < 4; ++nt) { b1v[nt] = b1[nt * 16 + m]; wrv[nt] = Wr[nt * 16 + m]; }
    const float brv = br[0];
    const v8f zero8 = {0.f,0.f,0.f,0.f,0.f,0.f,0.f,0.f};

    for (int ri = 0; ri < 8; ++ri) {
        const int r = rbase + ri;
        const _Float16* hrrow = HRb + r * HDIM0;
        v8f acc[4] = {zero8, zero8, zero8, zero8};
#pragma unroll
        for (int kt = 0; kt < 4; ++kt) {
            int k0 = 32 * kt + 8 * hi;
            v16h hv = ld16(hrrow, k0, k0 + 16);
            v16h a;
#pragma unroll
            for (int i = 0; i < 16; ++i) {
                _Float16 x = hlv[kt][i] + hv[i];          // v_pk_add_f16
                a[i] = x > (_Float16)0 ? x : (_Float16)0; // relu (v_pk_max)
            }
#pragma unroll
            for (int nt = 0; nt < 4; ++nt)
                acc[nt] = __builtin_amdgcn_wmma_f32_16x16x32_f16(
                    false, a, false, bfrag[kt][nt], (short)0, acc[nt], false, false);
        }
        // Dense(1): relu(acc + b1) . Wr over n = lane&15 across 4 n-tiles.
        float psum[8];
#pragma unroll
        for (int i = 0; i < 8; ++i) psum[i] = 0.f;
#pragma unroll
        for (int nt = 0; nt < 4; ++nt)
#pragma unroll
            for (int i = 0; i < 8; ++i) {
                float v = acc[nt][i] + b1v[nt];
                v = v > 0.f ? v : 0.f;
                psum[i] = fmaf(v, wrv[nt], psum[i]);
            }
        // Reduce over the 16 n-lanes of each half-wave.
#pragma unroll
        for (int mask = 8; mask >= 1; mask >>= 1)
#pragma unroll
            for (int i = 0; i < 8; ++i)
                psum[i] += __shfl_xor(psum[i], mask, 32);
        if (m == 0) {
#pragma unroll
            for (int i = 0; i < 8; ++i) {
                int l = l0 + i + 8 * hi;       // C layout: VGPR i -> M = i + 8*hi
                float x = psum[i] + brv;
                float e = x > 0.f ? x : (expf(x) - 1.f);   // elu
                Aout[l * NROW + r] = e;
                if (transposeF16) Af16[r * NROW + l] = (_Float16)e;
                else              Af16[l * NROW + r] = (_Float16)e;
            }
        }
    }
}

// ---------------------------------------------------------------------------
// Stage 3: sigmoid(A_chg @ A_gd) with f16 WMMA. One 16x16 tile per wave,
// K = 1024 in 32-element WMMA steps; B operand read from transposed A_gd copy.
// ---------------------------------------------------------------------------
__global__ __launch_bounds__(256) void gemm_sigmoid_kernel(
    const _Float16* __restrict__ Af, const _Float16* __restrict__ BTf,
    float* __restrict__ Out)
{
    const int tid  = threadIdx.x;
    const int lane = tid & 31;
    const int wave = tid >> 5;
    const int m    = lane & 15;
    const int hi   = lane >> 4;
    const int tile = blockIdx.x * 8 + wave;   // 64x64 tiles
    const int ty = tile >> 6;                 // ch tile
    const int tx = tile & 63;                 // d tile

    const _Float16* arow = Af  + (ty * 16 + m) * NROW;
    const _Float16* brow = BTf + (tx * 16 + m) * NROW;
    v8f acc = {0.f,0.f,0.f,0.f,0.f,0.f,0.f,0.f};

    for (int kb = 0; kb < NROW; kb += 32) {
        int ka = kb + 8 * hi;
        v16h a = ld16(arow, ka, ka + 16);      // A frag: row-major contiguous runs
        int kc = kb + 8 * hi;
        v16h b = ld16(brow, kc, kc + 16);      // B frag: contiguous via transposed copy
        acc = __builtin_amdgcn_wmma_f32_16x16x32_f16(false, a, false, b, (short)0, acc, false, false);
    }
#pragma unroll
    for (int i = 0; i < 8; ++i) {
        int row = ty * 16 + i + 8 * hi;
        float x = acc[i];
        Out[row * NROW + tx * 16 + m] = 1.f / (1.f + expf(-x));
    }
}

// ---------------------------------------------------------------------------
extern "C" void kernel_launch(void* const* d_in, const int* in_sizes, int n_in,
                              void* d_out, int out_size, void* d_ws, size_t ws_size,
                              hipStream_t stream) {
    const float* X_ch   = (const float*)d_in[0];
    const float* X_g    = (const float*)d_in[1];
    const float* X_d    = (const float*)d_in[2];
    const float* W_chg0 = (const float*)d_in[3];
    const float* b_chg0 = (const float*)d_in[4];
    const float* W_chg1 = (const float*)d_in[5];
    const float* b_chg1 = (const float*)d_in[6];
    const float* W_chgr = (const float*)d_in[7];
    const float* b_chgr = (const float*)d_in[8];
    const float* W_dg0  = (const float*)d_in[9];
    const float* b_dg0  = (const float*)d_in[10];
    const float* W_dg1  = (const float*)d_in[11];
    const float* b_dg1  = (const float*)d_in[12];
    const float* W_dgr  = (const float*)d_in[13];
    const float* b_dgr  = (const float*)d_in[14];

    float* out      = (float*)d_out;
    float* out_sig  = out;                      // [1M] sigmoid(A_chg @ A_gd)
    float* out_Achg = out + 1048576;            // [1M] A_chg
    float* out_Agd  = out + 2097152;            // [1M] A_gd

    _Float16* ws = (_Float16*)d_ws;
    _Float16* HL_ch   = ws;                     // 1024*128
    _Float16* HRb_g   = ws + 131072;
    _Float16* HL_g    = ws + 262144;
    _Float16* HRb_d   = ws + 393216;
    _Float16* W1T_chg = ws + 524288;            // 64*128
    _Float16* W1T_dg  = ws + 532480;
    _Float16* Achg16  = ws + 540672;            // 1024*1024 row-major
    _Float16* AgdT16  = ws + 540672 + 1048576;  // 1024*1024 transposed (B operand)

    // Stage 1: input projections + b0 fold, and W1 transpose/convert.
    stage1_h_kernel<<<2048, 256, 0, stream>>>(X_ch, X_g, X_d, W_chg0, b_chg0,
                                              W_dg0, b_dg0, HL_ch, HRb_g, HL_g, HRb_d);
    stage1_w_kernel<<<64, 256, 0, stream>>>(W_chg1, W_dg1, W1T_chg, W1T_dg);

    // Stage 2: the two pairwise MLPs (dominant compute, WMMA).
    dim3 grid2(64, 16);
    pair_mlp_kernel<<<grid2, 256, 0, stream>>>(HL_ch, HRb_g, W1T_chg, b_chg1,
                                               W_chgr, b_chgr, out_Achg, Achg16, 0);
    pair_mlp_kernel<<<grid2, 256, 0, stream>>>(HL_g, HRb_d, W1T_dg, b_dg1,
                                               W_dgr, b_dgr, out_Agd, AgdT16, 1);

    // Stage 3: sigmoid(A_chg @ A_gd).
    gemm_sigmoid_kernel<<<512, 256, 0, stream>>>(Achg16, AgdT16, out_sig);
}
